// PointNet2Cls_70205535421275
// MI455X (gfx1250) — compile-verified, matrix operand order checked by reading, not run on hardware
//
#include <hip/hip_runtime.h>
#include <hip/hip_bf16.h>
#include <math.h>

typedef __attribute__((ext_vector_type(16))) _Float16 v16h;
typedef __attribute__((ext_vector_type(8)))  _Float16 v8h;
typedef __attribute__((ext_vector_type(8)))  float    v8f;

// ---------------------------------------------------------------------------
// Split (B,6,N) channel-major input into coords f32 (B,N,3) and normals f16.
// ---------------------------------------------------------------------------
__global__ void k_split(const float* __restrict__ xyz, float* __restrict__ coords,
                        _Float16* __restrict__ normh, int B, int N) {
    int idx = blockIdx.x * blockDim.x + threadIdx.x;
    if (idx >= B * N) return;
    int b = idx / N, n = idx % N;
    const float* src = xyz + (size_t)b * 6 * N + n;
    size_t o = ((size_t)b * N + n) * 3;
    coords[o + 0] = src[0 * (size_t)N];
    coords[o + 1] = src[1 * (size_t)N];
    coords[o + 2] = src[2 * (size_t)N];
    normh[o + 0] = (_Float16)src[3 * (size_t)N];
    normh[o + 1] = (_Float16)src[4 * (size_t)N];
    normh[o + 2] = (_Float16)src[5 * (size_t)N];
}

// ---------------------------------------------------------------------------
// Farthest point sampling: one block per batch, running min-distance array
// kept in LDS (N <= 4096 -> 16KB of the 320KB/WGP). Emits the incoming `far`
// each iteration (matches the lax.scan output), argmax ties -> lowest index.
// ---------------------------------------------------------------------------
__global__ __launch_bounds__(256) void k_fps(const float* __restrict__ xyz,
                                             int* __restrict__ fidx, int N, int npoint) {
    int b = blockIdx.x;
    const float* P = xyz + (size_t)b * N * 3;
    __shared__ float D[4096];
    __shared__ float sv[256];
    __shared__ int   si[256];
    int t = threadIdx.x;
    for (int i = t; i < N; i += blockDim.x) D[i] = 1e10f;
    __syncthreads();
    int far = 0;
    for (int it = 0; it < npoint; ++it) {
        if (t == 0) fidx[(size_t)b * npoint + it] = far;
        float cx = P[(size_t)far * 3 + 0];
        float cy = P[(size_t)far * 3 + 1];
        float cz = P[(size_t)far * 3 + 2];
        float lm = -1.0f; int li = 0x7fffffff;
        for (int i = t; i < N; i += blockDim.x) {
            float dx = P[(size_t)i * 3 + 0] - cx;
            float dy = P[(size_t)i * 3 + 1] - cy;
            float dz = P[(size_t)i * 3 + 2] - cz;
            float d = fminf(D[i], dx * dx + dy * dy + dz * dz);
            D[i] = d;
            if (d > lm) { lm = d; li = i; }
        }
        sv[t] = lm; si[t] = li;
        __syncthreads();
        for (int off = 128; off > 0; off >>= 1) {
            if (t < off) {
                if (sv[t + off] > sv[t] ||
                    (sv[t + off] == sv[t] && si[t + off] < si[t])) {
                    sv[t] = sv[t + off]; si[t] = si[t + off];
                }
            }
            __syncthreads();
        }
        far = si[0];
        __syncthreads();
    }
}

// ---------------------------------------------------------------------------
// Gather selected centroids.
// ---------------------------------------------------------------------------
__global__ void k_gather_xyz(const float* __restrict__ xyz, const int* __restrict__ fidx,
                             float* __restrict__ newxyz, int B, int N, int S) {
    int idx = blockIdx.x * blockDim.x + threadIdx.x;
    if (idx >= B * S) return;
    int b = idx / S, s = idx % S;
    int gi = fidx[(size_t)b * S + s];
    const float* p = xyz + ((size_t)b * N + gi) * 3;
    float* o = newxyz + ((size_t)b * S + s) * 3;
    o[0] = p[0]; o[1] = p[1]; o[2] = p[2];
}

// ---------------------------------------------------------------------------
// Ball query: ascending-index hits, pad with first hit (reference semantics).
// ---------------------------------------------------------------------------
__global__ void k_ball(const float* __restrict__ xyz, const float* __restrict__ newxyz,
                       int* __restrict__ gidx, int B, int N, int S, int nsample, float r2) {
    int idx = blockIdx.x * blockDim.x + threadIdx.x;
    if (idx >= B * S) return;
    int b = idx / S, s = idx % S;
    const float* P = xyz + (size_t)b * N * 3;
    const float* q = newxyz + ((size_t)b * S + s) * 3;
    int* out = gidx + ((size_t)b * S + s) * nsample;
    float qx = q[0], qy = q[1], qz = q[2];
    int cnt = 0, first = 0;
    for (int i = 0; i < N && cnt < nsample; ++i) {
        float dx = P[(size_t)i * 3 + 0] - qx;
        float dy = P[(size_t)i * 3 + 1] - qy;
        float dz = P[(size_t)i * 3 + 2] - qz;
        if (dx * dx + dy * dy + dz * dz <= r2) {
            if (cnt == 0) first = i;
            out[cnt++] = i;
        }
    }
    for (; cnt < nsample; ++cnt) out[cnt] = first;
}

// ---------------------------------------------------------------------------
// Group: gather neighbor coords (centered) + f16 features, zero-pad to Cpad.
// Output rows (b,s,k) are WMMA A-matrix rows (f16, stride Cpad).
// ---------------------------------------------------------------------------
__global__ void k_group(const float* __restrict__ xyz, const _Float16* __restrict__ feats,
                        const float* __restrict__ newxyz, const int* __restrict__ gidx,
                        _Float16* __restrict__ A, int B, int N, int S, int K, int Cf, int Cpad) {
    int idx = blockIdx.x * blockDim.x + threadIdx.x;
    if (idx >= B * S * K) return;
    int bs = idx / K;
    int b = bs / S;
    int gi = gidx[(size_t)idx];
    const float* p = xyz + ((size_t)b * N + gi) * 3;
    const float* q = newxyz + (size_t)bs * 3;
    _Float16* o = A + (size_t)idx * Cpad;
    o[0] = (_Float16)(p[0] - q[0]);
    o[1] = (_Float16)(p[1] - q[1]);
    o[2] = (_Float16)(p[2] - q[2]);
    const _Float16* f = feats + ((size_t)b * N + gi) * Cf;
    for (int c = 0; c < Cf; ++c) o[3 + c] = f[c];
    for (int c = 3 + Cf; c < Cpad; ++c) o[c] = (_Float16)0.f;
}

// ---------------------------------------------------------------------------
// group_all for SA3: rows (b,k) = [l2_xyz(3) | l2_feats(Cf)] zero-padded.
// ---------------------------------------------------------------------------
__global__ void k_build_sa3(const float* __restrict__ xyz, const _Float16* __restrict__ feats,
                            _Float16* __restrict__ A, int total, int Cf, int Cpad) {
    int idx = blockIdx.x * blockDim.x + threadIdx.x;
    if (idx >= total) return;
    const float* p = xyz + (size_t)idx * 3;
    const _Float16* f = feats + (size_t)idx * Cf;
    _Float16* o = A + (size_t)idx * Cpad;
    o[0] = (_Float16)p[0]; o[1] = (_Float16)p[1]; o[2] = (_Float16)p[2];
    for (int c = 0; c < Cf; ++c) o[3 + c] = f[c];
    for (int c = 3 + Cf; c < Cpad; ++c) o[c] = (_Float16)0.f;
}

// ---------------------------------------------------------------------------
// Pad + convert weight (Cout,Cin) f32 -> (Coutpad,Cinpad) f16 row-major.
// ---------------------------------------------------------------------------
__global__ void k_pad_weight(const float* __restrict__ W, _Float16* __restrict__ Wp,
                             int Cout, int Cin, int Coutpad, int Cinpad) {
    int idx = blockIdx.x * blockDim.x + threadIdx.x;
    if (idx >= Coutpad * Cinpad) return;
    int r = idx / Cinpad, c = idx % Cinpad;
    float v = (r < Cout && c < Cin) ? W[(size_t)r * Cin + c] : 0.f;
    Wp[idx] = (_Float16)v;
}

// ---------------------------------------------------------------------------
// WMMA GEMM: out[M,Npad] = A[M,Kpad] * Wp[Npad,Kpad]^T with BN+ReLU epilogue.
// PRECONDITION: Kpad % 64 == 0 (host pads K dims accordingly).
// Copy-free ping-pong pipeline: two statically named fragment sets (A/B)
// alternate as the K loop advances by 64. wmma(setA) only has to wait for
// setA's loads while setB's reloads are still outstanding (in-order VMEM
// completion -> partial s_wait_loadcnt), so each wave hides a full load
// latency behind one WMMA group with zero rotation moves. Because Kpad is a
// multiple of 64, the steady-state loop needs no bounds guards and the last
// two WMMA groups are peeled after the loop.
// Fragment layout per CDNA5 ISA 7.12.2:
//   A/B lane L: row/col = L&15, hi = L>>4, element e -> K = (e>>3)*16 + hi*8 + (e&7)
//   => two contiguous 8-half (16B) loads per lane per 32-K step.
//   C/D lane L, vreg r: row = tm + hi*8 + r, col = tn + (L&15).
// ---------------------------------------------------------------------------
template <int NT>
__global__ __launch_bounds__(256) void k_wmma_gemm(
    const _Float16* __restrict__ A, const _Float16* __restrict__ Bw,
    const float* __restrict__ bias, const float* __restrict__ gmul,
    const float* __restrict__ badd, float scale, int relu, int Nvalid,
    _Float16* __restrict__ outH, float* __restrict__ outF,
    int M, int Npad, int Kpad) {
    const int lane = threadIdx.x & 31;
    const int wv = blockIdx.x * (blockDim.x >> 5) + (threadIdx.x >> 5);
    const int ng = (Npad >> 4) / NT;        // N-tile groups per M tile
    const int nt = (M >> 4) * ng;
    if (wv >= nt) return;
    const int tm = (wv / ng) << 4;
    const int tn0 = (wv % ng) * (NT << 4);
    const int hi = lane >> 4;
    const int l15 = lane & 15;

    const _Float16* arow = A + (size_t)(tm + l15) * Kpad + hi * 8;
    const _Float16* brow[NT];
#pragma unroll
    for (int j = 0; j < NT; ++j)
        brow[j] = Bw + (size_t)(tn0 + j * 16 + l15) * Kpad + hi * 8;

    v8f acc[NT];
#pragma unroll
    for (int j = 0; j < NT; ++j) acc[j] = (v8f){};

    // ---- prologue: fill both fragment sets (K steps 0 and 32) ----
    v8h a0A, a1A, b0A[NT], b1A[NT];
    v8h a0B, a1B, b0B[NT], b1B[NT];
    a0A = *(const v8h*)(arow);
    a1A = *(const v8h*)(arow + 16);
#pragma unroll
    for (int j = 0; j < NT; ++j) {
        b0A[j] = *(const v8h*)(brow[j]);
        b1A[j] = *(const v8h*)(brow[j] + 16);
    }
    a0B = *(const v8h*)(arow + 32);
    a1B = *(const v8h*)(arow + 48);
#pragma unroll
    for (int j = 0; j < NT; ++j) {
        b0B[j] = *(const v8h*)(brow[j] + 32);
        b1B[j] = *(const v8h*)(brow[j] + 48);
    }

    // ---- steady state: consume one set, reload it for k+64, alternate ----
    int k0 = 0;
    for (; k0 + 64 < Kpad; k0 += 64) {
        {
            v16h av = __builtin_shufflevector(a0A, a1A, 0,1,2,3,4,5,6,7,8,9,10,11,12,13,14,15);
#pragma unroll
            for (int j = 0; j < NT; ++j) {
                v16h bv = __builtin_shufflevector(b0A[j], b1A[j],
                                                  0,1,2,3,4,5,6,7,8,9,10,11,12,13,14,15);
                acc[j] = __builtin_amdgcn_wmma_f32_16x16x32_f16(
                    false, av, false, bv, (short)0, acc[j], false, false);
            }
            a0A = *(const v8h*)(arow + k0 + 64);
            a1A = *(const v8h*)(arow + k0 + 80);
#pragma unroll
            for (int j = 0; j < NT; ++j) {
                b0A[j] = *(const v8h*)(brow[j] + k0 + 64);
                b1A[j] = *(const v8h*)(brow[j] + k0 + 80);
            }
        }
        {
            v16h av = __builtin_shufflevector(a0B, a1B, 0,1,2,3,4,5,6,7,8,9,10,11,12,13,14,15);
#pragma unroll
            for (int j = 0; j < NT; ++j) {
                v16h bv = __builtin_shufflevector(b0B[j], b1B[j],
                                                  0,1,2,3,4,5,6,7,8,9,10,11,12,13,14,15);
                acc[j] = __builtin_amdgcn_wmma_f32_16x16x32_f16(
                    false, av, false, bv, (short)0, acc[j], false, false);
            }
            a0B = *(const v8h*)(arow + k0 + 96);
            a1B = *(const v8h*)(arow + k0 + 112);
#pragma unroll
            for (int j = 0; j < NT; ++j) {
                b0B[j] = *(const v8h*)(brow[j] + k0 + 96);
                b1B[j] = *(const v8h*)(brow[j] + k0 + 112);
            }
        }
    }
    // ---- drain: last two K steps ----
    {
        v16h av = __builtin_shufflevector(a0A, a1A, 0,1,2,3,4,5,6,7,8,9,10,11,12,13,14,15);
#pragma unroll
        for (int j = 0; j < NT; ++j) {
            v16h bv = __builtin_shufflevector(b0A[j], b1A[j],
                                              0,1,2,3,4,5,6,7,8,9,10,11,12,13,14,15);
            acc[j] = __builtin_amdgcn_wmma_f32_16x16x32_f16(
                false, av, false, bv, (short)0, acc[j], false, false);
        }
    }
    {
        v16h av = __builtin_shufflevector(a0B, a1B, 0,1,2,3,4,5,6,7,8,9,10,11,12,13,14,15);
#pragma unroll
        for (int j = 0; j < NT; ++j) {
            v16h bv = __builtin_shufflevector(b0B[j], b1B[j],
                                              0,1,2,3,4,5,6,7,8,9,10,11,12,13,14,15);
            acc[j] = __builtin_amdgcn_wmma_f32_16x16x32_f16(
                false, av, false, bv, (short)0, acc[j], false, false);
        }
    }

#pragma unroll
    for (int j = 0; j < NT; ++j) {
        const int col = tn0 + j * 16 + l15;
        const bool valid = col < Nvalid;
        float bi = 0.f, gg = 1.f, bb = 0.f;
        if (valid) {
            if (bias) bi = bias[col];
            if (gmul) gg = gmul[col];
            if (badd) bb = badd[col];
        }
#pragma unroll
        for (int r = 0; r < 8; ++r) {
            int row = tm + hi * 8 + r;
            float v = valid ? ((acc[j][r] + bi) * scale * gg + bb) : 0.f;
            if (relu) v = fmaxf(v, 0.f);
            size_t o = (size_t)row * Npad + col;
            if (outH) outH[o] = (_Float16)v;
            if (outF) outF[o] = v;
        }
    }
}

// ---------------------------------------------------------------------------
// Max-pool over the nsample axis: in (Stot,K,C) f16 -> (Stot,C).
// ---------------------------------------------------------------------------
__global__ void k_maxpool(const _Float16* __restrict__ in, _Float16* __restrict__ outH,
                          float* __restrict__ outF, int Stot, int K, int C) {
    int idx = blockIdx.x * blockDim.x + threadIdx.x;
    if (idx >= Stot * C) return;
    int s = idx / C, c = idx % C;
    const _Float16* p = in + (size_t)s * K * C + c;
    float m = (float)p[0];
    for (int k = 1; k < K; ++k) m = fmaxf(m, (float)p[(size_t)k * C + c]);
    if (outH) outH[idx] = (_Float16)m;
    if (outF) outF[idx] = m;
}

// ---------------------------------------------------------------------------
// Row-wise log_softmax over the first C columns (ld = padded width).
// ---------------------------------------------------------------------------
__global__ void k_logsoftmax(const float* __restrict__ logits, float* __restrict__ out,
                             int B, int C, int ld) {
    int b = blockIdx.x * blockDim.x + threadIdx.x;
    if (b >= B) return;
    const float* r = logits + (size_t)b * ld;
    float m = r[0];
    for (int c = 1; c < C; ++c) m = fmaxf(m, r[c]);
    float sum = 0.f;
    for (int c = 0; c < C; ++c) sum += expf(r[c] - m);
    float lse = m + logf(sum);
    for (int c = 0; c < C; ++c) out[(size_t)b * C + c] = r[c] - lse;
}

// ---------------------------------------------------------------------------
// Host launcher
// ---------------------------------------------------------------------------
extern "C" void kernel_launch(void* const* d_in, const int* in_sizes, int n_in,
                              void* d_out, int out_size, void* d_ws, size_t ws_size,
                              hipStream_t stream) {
    (void)in_sizes; (void)n_in; (void)out_size; (void)ws_size;

    const int B = 32, N0 = 4096;
    const int S1 = 512, K1 = 32;
    const int S2 = 128, K2 = 64;
    const float R1 = 0.2f, R2 = 0.4f;
    const float SNORM = 1.0f / sqrtf(1.0f + 1e-5f);
    // K paddings (all multiples of 64 for the ping-pong GEMM)
    const int CP1 = 64, CP2 = 192, CP3 = 320;

    // ---- input pointers (setup_inputs flattening order) ----
    const float* xyz = (const float*)d_in[0];
    const float* saW[3][3]; const float* saB[3][3]; const float* saG[3][3]; const float* saT[3][3];
    for (int s = 0; s < 3; ++s)
        for (int l = 0; l < 3; ++l) {
            int base = 1 + s * 12 + l * 4;
            saW[s][l] = (const float*)d_in[base + 0];
            saB[s][l] = (const float*)d_in[base + 1];
            saG[s][l] = (const float*)d_in[base + 2];
            saT[s][l] = (const float*)d_in[base + 3];
        }
    const float* hw1 = (const float*)d_in[37]; const float* hb1 = (const float*)d_in[38];
    const float* hg1 = (const float*)d_in[39]; const float* ht1 = (const float*)d_in[40];
    const float* hw2 = (const float*)d_in[41]; const float* hb2 = (const float*)d_in[42];
    const float* hg2 = (const float*)d_in[43]; const float* ht2 = (const float*)d_in[44];
    const float* hw3 = (const float*)d_in[45]; const float* hb3 = (const float*)d_in[46];

    // ---- workspace carve-out ----
    size_t off = 0;
    auto alloc = [&](size_t bytes) -> void* {
        off = (off + 255) & ~(size_t)255;
        void* p = (char*)d_ws + off;
        off += bytes;
        return p;
    };
    float*    coords  = (float*)   alloc((size_t)B * N0 * 3 * 4);
    _Float16* normh   = (_Float16*)alloc((size_t)B * N0 * 3 * 2);
    int*      fidx1   = (int*)     alloc((size_t)B * S1 * 4);
    float*    newxyz1 = (float*)   alloc((size_t)B * S1 * 3 * 4);
    int*      gidx1   = (int*)     alloc((size_t)B * S1 * K1 * 4);
    int*      fidx2   = (int*)     alloc((size_t)B * S2 * 4);
    float*    newxyz2 = (float*)   alloc((size_t)B * S2 * 3 * 4);
    int*      gidx2   = (int*)     alloc((size_t)B * S2 * K2 * 4);
    _Float16* l1h     = (_Float16*)alloc((size_t)B * S1 * 128 * 2);
    _Float16* l2h     = (_Float16*)alloc((size_t)B * S2 * 256 * 2);
    _Float16* l3h     = (_Float16*)alloc((size_t)B * 1024 * 2);
    _Float16* hact1   = (_Float16*)alloc((size_t)B * 512 * 2);
    _Float16* hact2   = (_Float16*)alloc((size_t)B * 256 * 2);
    float*    logits  = (float*)   alloc((size_t)B * 48 * 4);
    // padded weights (f16), Cin padded to multiples of 64
    _Float16* wp[12];
    const int wco[12] = { 64, 64, 128, 128, 128, 256, 256, 512, 1024, 512, 256, 48 };
    const int wci[12] = { CP1, 64, 64, CP2, 128, 128, CP3, 256, 512, 1024, 512, 256 };
    for (int i = 0; i < 12; ++i) wp[i] = (_Float16*)alloc((size_t)wco[i] * wci[i] * 2);
    // ping-pong activation buffers
    // bufA max: SA2 A = 262144*192 = 50.3M halves; bufB max: SA1 L3 = 67.1M halves
    _Float16* bufA = (_Float16*)alloc((size_t)52 * 1024 * 1024 * 2);
    _Float16* bufB = (_Float16*)alloc((size_t)68 * 1024 * 1024 * 2);

    auto g1 = [&](int n) { return dim3((n + 255) / 256); };
    const dim3 TB(256);

    auto gemm = [&](const _Float16* Ap, const _Float16* Wp, const float* bias,
                    const float* gm, const float* bt, float scale, int relu, int Nvalid,
                    _Float16* oH, float* oF, int M, int Npad, int Kpad) {
        int ntN = Npad / 16;
        if (ntN % 4 == 0) {
            int tiles = (M / 16) * (ntN / 4);
            k_wmma_gemm<4><<<dim3((tiles + 7) / 8), TB, 0, stream>>>(
                Ap, Wp, bias, gm, bt, scale, relu, Nvalid, oH, oF, M, Npad, Kpad);
        } else if (ntN % 3 == 0) {
            int tiles = (M / 16) * (ntN / 3);
            k_wmma_gemm<3><<<dim3((tiles + 7) / 8), TB, 0, stream>>>(
                Ap, Wp, bias, gm, bt, scale, relu, Nvalid, oH, oF, M, Npad, Kpad);
        } else {
            int tiles = (M / 16) * ntN;
            k_wmma_gemm<1><<<dim3((tiles + 7) / 8), TB, 0, stream>>>(
                Ap, Wp, bias, gm, bt, scale, relu, Nvalid, oH, oF, M, Npad, Kpad);
        }
    };

    // ---- pad all weights once (cheap, deterministic) ----
    const float* wsrc[12] = { saW[0][0], saW[0][1], saW[0][2],
                              saW[1][0], saW[1][1], saW[1][2],
                              saW[2][0], saW[2][1], saW[2][2],
                              hw1, hw2, hw3 };
    const int wcoR[12] = { 64, 64, 128, 128, 128, 256, 256, 512, 1024, 512, 256, 40 };
    const int wciR[12] = {  6, 64,  64, 131, 128, 128, 259, 256,  512, 1024, 512, 256 };
    for (int i = 0; i < 12; ++i) {
        int n = wco[i] * wci[i];
        k_pad_weight<<<g1(n), TB, 0, stream>>>(wsrc[i], wp[i], wcoR[i], wciR[i], wco[i], wci[i]);
    }

    // ---- stage 0: split input ----
    k_split<<<g1(B * N0), TB, 0, stream>>>(xyz, coords, normh, B, N0);

    // ---- SA1 ----
    k_fps<<<dim3(B), TB, 0, stream>>>(coords, fidx1, N0, S1);
    k_gather_xyz<<<g1(B * S1), TB, 0, stream>>>(coords, fidx1, newxyz1, B, N0, S1);
    k_ball<<<g1(B * S1), TB, 0, stream>>>(coords, newxyz1, gidx1, B, N0, S1, K1, R1 * R1);
    k_group<<<g1(B * S1 * K1), TB, 0, stream>>>(coords, normh, newxyz1, gidx1,
                                                bufA, B, N0, S1, K1, 3, CP1);
    {
        int M = B * S1 * K1; // 524288
        gemm(bufA, wp[0], saB[0][0], saG[0][0], saT[0][0], SNORM, 1, 64, bufB, nullptr, M, 64, CP1);
        gemm(bufB, wp[1], saB[0][1], saG[0][1], saT[0][1], SNORM, 1, 64, bufA, nullptr, M, 64, 64);
        gemm(bufA, wp[2], saB[0][2], saG[0][2], saT[0][2], SNORM, 1, 128, bufB, nullptr, M, 128, 64);
        k_maxpool<<<g1(B * S1 * 128), TB, 0, stream>>>(bufB, l1h, nullptr, B * S1, K1, 128);
    }

    // ---- SA2 ----
    k_fps<<<dim3(B), TB, 0, stream>>>(newxyz1, fidx2, S1, S2);
    k_gather_xyz<<<g1(B * S2), TB, 0, stream>>>(newxyz1, fidx2, newxyz2, B, S1, S2);
    k_ball<<<g1(B * S2), TB, 0, stream>>>(newxyz1, newxyz2, gidx2, B, S1, S2, K2, R2 * R2);
    k_group<<<g1(B * S2 * K2), TB, 0, stream>>>(newxyz1, l1h, newxyz2, gidx2,
                                                bufA, B, S1, S2, K2, 128, CP2);
    {
        int M = B * S2 * K2; // 262144
        gemm(bufA, wp[3], saB[1][0], saG[1][0], saT[1][0], SNORM, 1, 128, bufB, nullptr, M, 128, CP2);
        gemm(bufB, wp[4], saB[1][1], saG[1][1], saT[1][1], SNORM, 1, 128, bufA, nullptr, M, 128, 128);
        gemm(bufA, wp[5], saB[1][2], saG[1][2], saT[1][2], SNORM, 1, 256, bufB, nullptr, M, 256, 128);
        k_maxpool<<<g1(B * S2 * 256), TB, 0, stream>>>(bufB, l2h, nullptr, B * S2, K2, 256);
    }

    // ---- SA3 (group_all) ----
    k_build_sa3<<<g1(B * S2), TB, 0, stream>>>(newxyz2, l2h, bufA, B * S2, 256, CP3);
    {
        int M = B * S2; // 4096 pixels, grouped as (B, 1, 128)
        gemm(bufA, wp[6], saB[2][0], saG[2][0], saT[2][0], SNORM, 1, 256, bufB, nullptr, M, 256, CP3);
        gemm(bufB, wp[7], saB[2][1], saG[2][1], saT[2][1], SNORM, 1, 512, bufA, nullptr, M, 512, 256);
        gemm(bufA, wp[8], saB[2][2], saG[2][2], saT[2][2], SNORM, 1, 1024, bufB, nullptr, M, 1024, 512);
        // l3_pts: f16 for head, f32 straight into second tuple output d_out[1280:]
        k_maxpool<<<g1(B * 1024), TB, 0, stream>>>(bufB, l3h, (float*)d_out + 1280,
                                                   B, S2, 1024);
    }

    // ---- classification head ----
    gemm(l3h,   wp[9],  hb1, hg1, ht1, SNORM, 1, 512, hact1, nullptr, B, 512, 1024);
    gemm(hact1, wp[10], hb2, hg2, ht2, SNORM, 1, 256, hact2, nullptr, B, 256, 512);
    gemm(hact2, wp[11], hb3, nullptr, nullptr, 1.0f, 0, 40, nullptr, logits, B, 48, 256);
    k_logsoftmax<<<g1(B), TB, 0, stream>>>(logits, (float*)d_out, B, 40, 48);
}
